// myLSTMGroupCell_82514911691162
// MI455X (gfx1250) — matrix operand order repaired and checked by reading, not compiled, not run on hardware
//
#include <hip/hip_runtime.h>
#include <hip/hip_bf16.h>
#include <math.h>

typedef __bf16 bf16_t;
typedef __attribute__((ext_vector_type(16))) __bf16 v16bf;
typedef __attribute__((ext_vector_type(8)))  __bf16 v8bf;
typedef __attribute__((ext_vector_type(4)))  __bf16 v4bf;
typedef __attribute__((ext_vector_type(8)))  float  v8f;

#define B_DIM   8192
#define I_DIM   1024
#define H_DIM   2048
#define HG      1024
#define WR_DIM  256
#define ACAT_LD 768   // 256 (xw) + 2 groups * 2 idx * 128 (t)

// ---- fragment loaders -----------------------------------------------------
// A (16x32 bf16): p must already point at row*lda + (lane>>4)*8.
// elements j=0..7 at p[kOff + j], j=8..15 at p[kOff + 16 + j-8].
__device__ __forceinline__ v16bf ld_a(const bf16_t* __restrict__ p, int kOff) {
  v8bf lo = *(const v8bf*)(p + kOff);
  v8bf hi = *(const v8bf*)(p + kOff + 16);
  v16bf r;
#pragma unroll
  for (int j = 0; j < 8; ++j) { r[j] = lo[j]; r[j + 8] = hi[j]; }
  return r;
}

// B (32x16 bf16) pre-packed: p points at tileBase + lane*16; tile stride 512.
__device__ __forceinline__ v16bf ld_b(const bf16_t* __restrict__ p, int elemOff) {
  v8bf lo = *(const v8bf*)(p + elemOff);
  v8bf hi = *(const v8bf*)(p + elemOff + 8);
  v16bf r;
#pragma unroll
  for (int j = 0; j < 8; ++j) { r[j] = lo[j]; r[j + 8] = hi[j]; }
  return r;
}

#define WMMA_BF16(a, b, c) \
  __builtin_amdgcn_wmma_f32_16x16x32_bf16(false, (a), false, (b), (short)0, (c), false, false)

// ---- f32 -> bf16 conversion (vectorized x4) -------------------------------
__global__ __launch_bounds__(256) void cvt_f32_bf16_kernel(
    const float4* __restrict__ src, v4bf* __restrict__ dst, int n4) {
  int i = blockIdx.x * blockDim.x + threadIdx.x;
  if (i >= n4) return;
  float4 v = src[i];
  v4bf o;
  o[0] = (bf16_t)v.x; o[1] = (bf16_t)v.y; o[2] = (bf16_t)v.z; o[3] = (bf16_t)v.w;
  dst[i] = o;
}

// ---- pack a K-major f32 matrix (element (k,n) at src[k*srcStride+n]) into
// the per-lane WMMA B-fragment layout, as bf16.
__global__ __launch_bounds__(256) void pack_b_bf16_kernel(
    const float* __restrict__ src, int srcStride, int K, int N,
    bf16_t* __restrict__ dst, int ktOff, int KTtot) {
  int tid = blockIdx.x * blockDim.x + threadIdx.x;
  if (tid >= K * N) return;
  int k = tid / N, n = tid - k * N;
  int kt = k >> 5, kin = k & 31;
  int nt = n >> 4, nin = n & 15;
  int lane = (((kin >> 3) & 1) << 4) | nin;  // B: lanes 0-15 K=0..7,16..23, lanes 16-31 K=8..15,24..31
  int j = (kin & 7) + ((kin >> 4) << 3);
  dst[((size_t)(nt * KTtot + ktOff + kt) * 32 + lane) * 16 + j] =
      (bf16_t)src[(size_t)k * srcStride + n];
}

// ---- generic bf16 GEMM (K=1024 fixed): one wave -> 32x16 tile, pipelined --
__global__ __launch_bounds__(256) void gemm_bf16_wmma_kernel(
    const bf16_t* __restrict__ A, int lda, const bf16_t* __restrict__ Bp,
    int Ntiles, bf16_t* __restrict__ out, int ldo, int colOff) {
  const int KT = 32;                 // 1024 / 32
  int wave = blockIdx.x * (blockDim.x >> 5) + (threadIdx.x >> 5);
  int lane = threadIdx.x & 31;
  int mt2 = wave / Ntiles;           // 32-row block index
  int nt  = wave - mt2 * Ntiles;
  int row0 = (mt2 << 5) + (lane & 15);
  const bf16_t* pa0 = A + (size_t)row0 * lda + ((lane >> 4) << 3);
  const bf16_t* pa1 = pa0 + (size_t)16 * lda;
  const bf16_t* pb  = Bp + (size_t)(nt * KT) * 512 + lane * 16;

  v8f acc0 = {}, acc1 = {};
  // software pipeline: fragments for kt+1 loaded before WMMAs of kt
  v16bf b  = ld_b(pb, 0);
  v16bf a0 = ld_a(pa0, 0);
  v16bf a1 = ld_a(pa1, 0);
#pragma unroll 4
  for (int kt = 0; kt < KT; ++kt) {
    int nk = (kt + 1 < KT) ? kt + 1 : kt;
    v16bf bn  = ld_b(pb, nk * 512);
    v16bf a0n = ld_a(pa0, nk * 32);
    v16bf a1n = ld_a(pa1, nk * 32);
    acc0 = WMMA_BF16(a0, b, acc0);
    acc1 = WMMA_BF16(a1, b, acc1);
    b = bn; a0 = a0n; a1 = a1n;
  }
  int col = (nt << 4) + (lane & 15) + colOff;
  int rbase = (mt2 << 5) + ((lane >> 4) << 3);
#pragma unroll
  for (int v = 0; v < 8; ++v) {
    out[(size_t)(rbase + v) * ldo + col]      = (bf16_t)acc0[v];
    out[(size_t)(rbase + 16 + v) * ldo + col] = (bf16_t)acc1[v];
  }
}

// ---- fused final stage: K=512 GEMM x 4 gates x 2 M-tiles + LSTM epilogue --
// Acat: (B x 768) bf16 = [xw(256) | t00(128) | t10(128) | t01(128) | t11(128)]
// Bcat: per (gate f, group g): packed (512 x 1024), KTtot=16
__global__ __launch_bounds__(256) void lstm_fused_gates_kernel(
    const bf16_t* __restrict__ Acat, const bf16_t* __restrict__ Bcat,
    const float* __restrict__ biases, const float* __restrict__ cprev,
    float* __restrict__ out) {
  int wave = blockIdx.x * (blockDim.x >> 5) + (threadIdx.x >> 5);
  int lane = threadIdx.x & 31;
  int mt2 = wave >> 7;     // 32-row block: 0..255
  int ht  = wave & 127;    // H tile: 0..127
  int g   = ht >> 6;       // group
  int ntl = ht & 63;       // N tile within group

  int row0 = (mt2 << 5) + (lane & 15);
  const bf16_t* paX0 = Acat + (size_t)row0 * ACAT_LD + ((lane >> 4) << 3);
  const bf16_t* paX1 = paX0 + (size_t)16 * ACAT_LD;
  const bf16_t* paT0 = paX0 + 256 + (g << 8);   // this group's t regions
  const bf16_t* paT1 = paT0 + (size_t)16 * ACAT_LD;

  const size_t BSZ = (size_t)512 * 1024;
  size_t bbase = (size_t)(ntl * 16) * 512 + lane * 16;
  const bf16_t* pb0 = Bcat + (size_t)(0 * 2 + g) * BSZ + bbase;
  const bf16_t* pb1 = Bcat + (size_t)(1 * 2 + g) * BSZ + bbase;
  const bf16_t* pb2 = Bcat + (size_t)(2 * 2 + g) * BSZ + bbase;
  const bf16_t* pb3 = Bcat + (size_t)(3 * 2 + g) * BSZ + bbase;

  v8f aF0 = {}, aI0 = {}, aO0 = {}, aG0 = {};
  v8f aF1 = {}, aI1 = {}, aO1 = {}, aG1 = {};

  // software pipeline across the whole K=512 (16 k-tiles; region switch at 8)
  v16bf a0 = ld_a(paX0, 0);
  v16bf a1 = ld_a(paX1, 0);
  v16bf b0 = ld_b(pb0, 0), b1 = ld_b(pb1, 0), b2 = ld_b(pb2, 0), b3 = ld_b(pb3, 0);
#pragma unroll
  for (int kt = 0; kt < 16; ++kt) {
    int nk = (kt + 1 < 16) ? kt + 1 : kt;            // compile-time per unroll
    const bf16_t* nA0 = (nk < 8) ? paX0 : paT0;
    const bf16_t* nA1 = (nk < 8) ? paX1 : paT1;
    int nao = (nk & 7) * 32;
    int nbo = nk * 512;
    v16bf na0 = ld_a(nA0, nao);
    v16bf na1 = ld_a(nA1, nao);
    v16bf nb0 = ld_b(pb0, nbo);
    v16bf nb1 = ld_b(pb1, nbo);
    v16bf nb2 = ld_b(pb2, nbo);
    v16bf nb3 = ld_b(pb3, nbo);
    aF0 = WMMA_BF16(a0, b0, aF0); aF1 = WMMA_BF16(a1, b0, aF1);
    aI0 = WMMA_BF16(a0, b1, aI0); aI1 = WMMA_BF16(a1, b1, aI1);
    aO0 = WMMA_BF16(a0, b2, aO0); aO1 = WMMA_BF16(a1, b2, aO1);
    aG0 = WMMA_BF16(a0, b3, aG0); aG1 = WMMA_BF16(a1, b3, aG1);
    a0 = na0; a1 = na1; b0 = nb0; b1 = nb1; b2 = nb2; b3 = nb3;
  }

  int hcol = (ht << 4) + (lane & 15);
  float bF = biases[0 * H_DIM + hcol];
  float bI = biases[1 * H_DIM + hcol];
  float bO = biases[2 * H_DIM + hcol];
  float bG = biases[3 * H_DIM + hcol];
  const size_t HOFF = (size_t)B_DIM * H_DIM;

#pragma unroll
  for (int half = 0; half < 2; ++half) {
    int rbase = (mt2 << 5) + (half << 4) + ((lane >> 4) << 3);
    v8f aF = half ? aF1 : aF0, aI = half ? aI1 : aI0;
    v8f aO = half ? aO1 : aO0, aG = half ? aG1 : aG0;
#pragma unroll
    for (int v = 0; v < 8; ++v) {
      size_t b = (size_t)(rbase + v);
      float pf = aF[v] + bF, pi = aI[v] + bI, po = aO[v] + bO, pg = aG[v] + bG;
      float fg = 1.0f / (1.0f + __expf(-pf));
      float ig = 1.0f / (1.0f + __expf(-pi));
      float og = 1.0f / (1.0f + __expf(-po));
      float gg = tanhf(pg);
      float cp = cprev[b * H_DIM + hcol];
      float cn = fg * cp + ig * gg;
      float hn = og * tanhf(cn);
      out[b * H_DIM + hcol]        = cn;   // c_next
      out[HOFF + b * H_DIM + hcol] = hn;   // h_next
    }
  }
}

// ---------------------------------------------------------------------------
extern "C" void kernel_launch(void* const* d_in, const int* in_sizes, int n_in,
                              void* d_out, int out_size, void* d_ws, size_t ws_size,
                              hipStream_t stream) {
  (void)in_sizes; (void)n_in; (void)out_size; (void)ws_size;
  const float* x  = (const float*)d_in[0];
  const float* h  = (const float*)d_in[1];
  const float* c  = (const float*)d_in[2];
  const float* W  = (const float*)d_in[3];
  const float* Ws = (const float*)d_in[4];
  const float* U0 = (const float*)d_in[5];
  const float* Ug = (const float*)d_in[6];
  const float* bi = (const float*)d_in[7];
  float* out = (float*)d_out;

  char* ws = (char*)d_ws;
  size_t off = 0;
  auto alloc = [&](size_t bytes) -> void* {
    void* p = ws + off;
    off += (bytes + 255) & ~(size_t)255;
    return p;
  };
  bf16_t* xbf  = (bf16_t*)alloc((size_t)B_DIM * I_DIM * 2);        // 16 MB
  bf16_t* hbf  = (bf16_t*)alloc((size_t)B_DIM * H_DIM * 2);        // 32 MB
  bf16_t* Acat = (bf16_t*)alloc((size_t)B_DIM * ACAT_LD * 2);      // 12 MB
  bf16_t* Wp   = (bf16_t*)alloc((size_t)I_DIM * WR_DIM * 2);       // 0.5 MB
  bf16_t* U0p  = (bf16_t*)alloc((size_t)4 * 1024 * 128 * 2);       // 1 MB
  bf16_t* Bcat = (bf16_t*)alloc((size_t)8 * 512 * 1024 * 2);       // 8 MB

  // 1) convert activations to bf16
  cvt_f32_bf16_kernel<<<(B_DIM * I_DIM / 4) / 256, 256, 0, stream>>>(
      (const float4*)x, (v4bf*)xbf, B_DIM * I_DIM / 4);
  cvt_f32_bf16_kernel<<<(B_DIM * H_DIM / 4) / 256, 256, 0, stream>>>(
      (const float4*)h, (v4bf*)hbf, B_DIM * H_DIM / 4);

  // 2) pack weights to WMMA B-fragment layout (bf16)
  pack_b_bf16_kernel<<<(I_DIM * WR_DIM) / 256, 256, 0, stream>>>(
      W, WR_DIM, I_DIM, WR_DIM, Wp, 0, 32);
  for (int idx = 0; idx < 2; ++idx)
    for (int g = 0; g < 2; ++g)
      pack_b_bf16_kernel<<<(1024 * 128) / 256, 256, 0, stream>>>(
          U0 + (size_t)(idx * 2 + g) * 1024 * 128, 128, 1024, 128,
          U0p + (size_t)(idx * 2 + g) * 1024 * 128, 0, 32);
  for (int f = 0; f < 4; ++f)
    for (int g = 0; g < 2; ++g) {
      bf16_t* dst = Bcat + (size_t)(f * 2 + g) * 512 * 1024;
      // rows 0..255: Ws[f][:, g*1024 + n]
      pack_b_bf16_kernel<<<(256 * 1024) / 256, 256, 0, stream>>>(
          Ws + (size_t)f * 256 * H_DIM + g * HG, H_DIM, 256, 1024, dst, 0, 16);
      // rows 256..383: Ugates[0][f][g]
      pack_b_bf16_kernel<<<(128 * 1024) / 256, 256, 0, stream>>>(
          Ug + (size_t)((0 * 4 + f) * 2 + g) * 128 * 1024, 1024, 128, 1024, dst, 8, 16);
      // rows 384..511: Ugates[1][f][g]
      pack_b_bf16_kernel<<<(128 * 1024) / 256, 256, 0, stream>>>(
          Ug + (size_t)((1 * 4 + f) * 2 + g) * 128 * 1024, 1024, 128, 1024, dst, 12, 16);
    }

  // 3) xw = x @ W  -> Acat cols [0,256)   (256 mt2-blocks x 16 nt = 4096 waves)
  gemm_bf16_wmma_kernel<<<(256 * 16) / 8, 256, 0, stream>>>(
      xbf, I_DIM, Wp, /*Ntiles=*/16, Acat, ACAT_LD, /*colOff=*/0);

  // 4) t[idx][g] = h[:, (g+idx)%2] @ U0[idx][g] -> Acat cols 256+g*256+idx*128
  for (int idx = 0; idx < 2; ++idx)
    for (int g = 0; g < 2; ++g) {
      int gsrc = (g + idx) & 1;
      gemm_bf16_wmma_kernel<<<(256 * 8) / 8, 256, 0, stream>>>(
          hbf + (size_t)gsrc * HG, H_DIM,
          U0p + (size_t)(idx * 2 + g) * 1024 * 128,
          /*Ntiles=*/8, Acat, ACAT_LD,
          /*colOff=*/256 + g * 256 + idx * 128);
    }

  // 5) fused 4-gate K=512 GEMM + LSTM epilogue (256 mt2 x 128 ht = 32768 waves)
  lstm_fused_gates_kernel<<<(256 * 128) / 8, 256, 0, stream>>>(
      Acat, Bcat, bi, c, out);
}
